// CtsConv_55448027791888
// MI455X (gfx1250) — matrix operand order repaired and checked by reading, not compiled, not run on hardware
//
#include <hip/hip_runtime.h>

typedef __attribute__((ext_vector_type(16))) _Float16 v16h;
typedef __attribute__((ext_vector_type(8)))  float    v8f;

#define B_    2
#define M_    192
#define N_    192
#define CIN   32
#define COUT  32
#define TAPS  125
#define RADIUS_INV 0.5f

// ---------------------------------------------------------------------------
// Prep: transpose fp32 kernel[o][i][tap] -> f16 kh[tap][o][i] in workspace.
// Makes every WMMA B-fragment a contiguous 16-half (32B) read per lane.
// ---------------------------------------------------------------------------
__global__ void cts_prep_kernel(const float* __restrict__ k4,
                                _Float16* __restrict__ kh) {
  int idx = blockIdx.x * blockDim.x + threadIdx.x;
  if (idx < TAPS * COUT * CIN) {
    int tap = idx / (COUT * CIN);
    int r   = idx % (COUT * CIN);
    int o   = r / CIN;
    int i   = r % CIN;
    kh[(tap * COUT + o) * CIN + i] = (_Float16)k4[(o * CIN + i) * TAPS + tap];
  }
}

// ---------------------------------------------------------------------------
// Main: one workgroup per (b, 16-row m tile). 8 wave32s.
// Stage 1: scatter att-weighted features onto the 125-tap grid in LDS (f32).
// Stage 2: [16 x 4000 x 32] GEMM via v_wmma_f32_16x16x32_f16, K-chunk = tap.
// ---------------------------------------------------------------------------
__global__ void __launch_bounds__(256)
cts_main_kernel(const float* __restrict__ field,
                const float* __restrict__ center,
                const float* __restrict__ feat,
                const float* __restrict__ mask,
                const _Float16* __restrict__ kh,
                float* __restrict__ out) {
  extern __shared__ float smem[];
  float* g   = smem;                    // [16][TAPS][CIN] f32 accum (256 KB)
  float* red = smem + 16 * TAPS * CIN;  // [8 waves][32 lanes][8] partial C

  const int tid  = threadIdx.x;
  const int wave = tid >> 5;
  const int lane = tid & 31;
  const int blk  = blockIdx.x;
  const int b    = blk / (M_ / 16);
  const int m0   = (blk % (M_ / 16)) * 16;

  for (int k = tid; k < 16 * TAPS * CIN; k += 256) g[k] = 0.0f;
  __syncthreads();

  const float kpi = 1.27323954473516f;  // 4/pi

  // ---- Stage 1: scatter. Wave w owns m-rows {w, w+8}; lane i owns channel i.
  for (int ml = wave; ml < 16; ml += 8) {
    const int m = m0 + ml;
    const float cx = center[(b * M_ + m) * 3 + 0];
    const float cy = center[(b * M_ + m) * 3 + 1];
    const float cz = center[(b * M_ + m) * 3 + 2];
    float* grow = g + ml * TAPS * CIN + lane;
    for (int n = 0; n < N_; ++n) {
      const float rx = (field[(b * N_ + n) * 3 + 0] - cx) * RADIUS_INV;
      const float ry = (field[(b * N_ + n) * 3 + 1] - cy) * RADIUS_INV;
      const float rz = (field[(b * N_ + n) * 3 + 2] - cz) * RADIUS_INV;
      const float r2 = rx * rx + ry * ry + rz * rz;
      if (r2 >= 1.0f) continue;  // relu((1-r2)^3) == 0 ; wave-uniform branch
      const float t   = 1.0f - r2;
      const float att = t * t * t * mask[b * N_ + n];
      const float fv  = feat[(b * N_ + n) * CIN + lane] * att;

      // map_polar_sqr
      const float r  = sqrtf(rx * rx + ry * ry + 1e-9f);
      const bool  c1 = (rx == 0.0f) && (ry == 0.0f);
      const bool  c2 = (fabsf(ry) <= fabsf(rx)) && !c1;
      const float xs = (rx == 0.0f) ? 1.0f : rx;
      const float ys = (ry == 0.0f) ? 1.0f : ry;
      const float sx = (rx > 0.0f) ? 1.0f : ((rx < 0.0f) ? -1.0f : 0.0f);
      const float sy = (ry > 0.0f) ? 1.0f : ((ry < 0.0f) ? -1.0f : 0.0f);
      const float x2 = sx * r;
      const float y2 = kpi * sx * r * atanf(ry / xs);
      const float x3 = kpi * sy * r * atanf(rx / ys);
      const float y3 = sy * r;
      const float gx = c1 ? 0.0f : (c2 ? x2 : x3);
      const float gy = c1 ? 0.0f : (c2 ? y2 : y3);
      const float gz = rz;

      // trilinear weights onto the 5x5x5 grid
      const float ix = ((gx + 1.0f) * 5.0f - 1.0f) * 0.5f;
      const float iy = ((gy + 1.0f) * 5.0f - 1.0f) * 0.5f;
      const float iz = ((gz + 1.0f) * 5.0f - 1.0f) * 0.5f;
      const float fx0 = floorf(ix), fy0 = floorf(iy), fz0 = floorf(iz);
      const float fxv = ix - fx0, fyv = iy - fy0, fzv = iz - fz0;
      const int x0 = (int)fx0, y0 = (int)fy0, z0 = (int)fz0;
#pragma unroll
      for (int c = 0; c < 8; ++c) {
        const int dx = c & 1, dy = (c >> 1) & 1, dz = (c >> 2) & 1;
        const int xc = x0 + dx, yc = y0 + dy, zc = z0 + dz;
        if (xc < 0 || xc > 4 || yc < 0 || yc > 4 || zc < 0 || zc > 4) continue;
        const float w = (dx ? fxv : 1.0f - fxv) *
                        (dy ? fyv : 1.0f - fyv) *
                        (dz ? fzv : 1.0f - fzv);
        const int tap = (zc * 5 + yc) * 5 + xc;
        grow[tap * CIN] += w * fv;  // lane-exclusive address: no atomics needed
      }
    }
  }
  __syncthreads();

  // ---- Stage 2: GEMM. wave = (N-tile) + 2*(K-partition of ~32 taps)
  const int tile = wave & 1;
  const int part = wave >> 1;
  const int tap0 = part * 32;
  const int tap1 = (part == 3) ? TAPS : tap0 + 32;
  const int mrow = lane & 15;
  const int hi   = lane >> 4;
  const int o0   = tile * 16;
  const int iA0  = hi * 16;       // A-frag K mapping: contiguous k = 16*hi + s (matches B)
  const int iA1  = hi * 16 + 8;

  v8f acc = {};
  // Unroll x2: two independent A/B fetch streams per iteration so the LDS
  // reads + f16 converts of one tap overlap the WMMA of the other.
#pragma unroll 2
  for (int tap = tap0; tap < tap1; ++tap) {
    const float* ga = g + (mrow * TAPS + tap) * CIN;
    v16h A;
#pragma unroll
    for (int j = 0; j < 8; ++j) A[j]     = (_Float16)ga[iA0 + j];
#pragma unroll
    for (int j = 0; j < 8; ++j) A[8 + j] = (_Float16)ga[iA1 + j];

    // B-frag: lanes 0-15 hold K=0..15, lanes 16-31 hold K=16..31 -> contiguous
    const _Float16* kb = kh + ((tap * COUT) + (o0 + mrow)) * CIN + hi * 16;
    v16h Bf;
#pragma unroll
    for (int j = 0; j < 16; ++j) Bf[j] = kb[j];

    acc = __builtin_amdgcn_wmma_f32_16x16x32_f16(
        /*neg_a=*/false, A, /*neg_b=*/false, Bf,
        /*c_mod=*/(short)0, acc, /*reuse_a=*/false, /*reuse_b=*/false);
  }

  // ---- cross-wave K reduction through LDS
  float* rslot = red + (wave * 32 + lane) * 8;
#pragma unroll
  for (int v = 0; v < 8; ++v) rslot[v] = acc[v];
  __syncthreads();

  if (wave < 2) {  // wave 0 -> N-tile 0 (partials at 0,2,4,6); wave 1 -> tile 1
#pragma unroll
    for (int v = 0; v < 8; ++v) {
      const float s = red[((wave + 0) * 32 + lane) * 8 + v] +
                      red[((wave + 2) * 32 + lane) * 8 + v] +
                      red[((wave + 4) * 32 + lane) * 8 + v] +
                      red[((wave + 6) * 32 + lane) * 8 + v];
      const int mr = m0 + v + hi * 8;     // C/D layout: M = v + 8*(lane>=16)
      const int o  = wave * 16 + mrow;    // N = lane % 16
      out[(b * M_ + mr) * COUT + o] = s;
    }
  }
}

extern "C" void kernel_launch(void* const* d_in, const int* in_sizes, int n_in,
                              void* d_out, int out_size, void* d_ws, size_t ws_size,
                              hipStream_t stream) {
  (void)in_sizes; (void)n_in; (void)out_size; (void)ws_size;
  const float* field  = (const float*)d_in[0];
  const float* center = (const float*)d_in[1];
  const float* feat   = (const float*)d_in[2];
  const float* mask   = (const float*)d_in[3];
  const float* kern   = (const float*)d_in[4];
  float* out = (float*)d_out;
  _Float16* kh = (_Float16*)d_ws;  // 125*32*32 f16 = 256000 B

  const int total = TAPS * COUT * CIN;
  cts_prep_kernel<<<(total + 255) / 256, 256, 0, stream>>>(kern, kh);

  const size_t lds = (size_t)(16 * TAPS * CIN + 8 * 32 * 8) * sizeof(float);
  cts_main_kernel<<<B_ * (M_ / 16), 256, lds, stream>>>(field, center, feat,
                                                        mask, kh, out);
}
